// InteractionLayer_19301583029076
// MI455X (gfx1250) — compile-verified, matrix operand order checked by reading, not run on hardware
//
#include <hip/hip_runtime.h>
#include <math.h>

typedef __attribute__((ext_vector_type(16))) _Float16 v16h;
typedef __attribute__((ext_vector_type(8)))  float    v8f;
typedef _Float16 half_t;

#define F  128
#define NB 8
#define NR 6
#define NS 42

// K index (even base of an f16 pair) held by VGPR-pair v for a lane half.
// A 16x32 f16 layout: lanes 0-15: v0..3 -> K{0,2,4,6}, v4..7 -> K{16,18,20,22};
// lanes 16-31: +8. B symmetric with N in place of M.
__device__ __forceinline__ int kmap(int v, int khalf) {
  return ((v < 4) ? (2 * v) : (8 + 2 * v)) + 8 * khalf;
}

// swish via hardware transcendentals: v_exp_f32 + v_rcp_f32 (no IEEE div expansion)
__device__ __forceinline__ float swishf(float x) {
  return x * __builtin_amdgcn_rcpf(1.0f + __expf(-x));
}

// Load a fragment from row-major f16 storage.
__device__ __forceinline__ v16h load_frag(const half_t* base, int row, int rowstride,
                                          int k0, int khalf) {
  v16h f;
#pragma unroll
  for (int v = 0; v < 8; ++v) {
    int k = k0 + kmap(v, khalf);
    f[2 * v]     = base[row * rowstride + k];
    f[2 * v + 1] = base[row * rowstride + k + 1];
  }
  return f;
}

// Async DMA one 16B chunk: global -> LDS (ASYNCcnt-tracked, no VGPR round trip).
__device__ __forceinline__ void async_copy16(void* ldsDst, const void* gSrc) {
  unsigned lds = (unsigned)(unsigned long long)ldsDst;          // low 32 bits = LDS offset
  unsigned long long g = (unsigned long long)gSrc;
  asm volatile("global_load_async_to_lds_b128 %0, %1, off"
               :: "v"(lds), "v"(g) : "memory");
}

__device__ __forceinline__ void wait_async0() {
  asm volatile("s_wait_asynccnt 0x0" ::: "memory");
}

// Stage one contiguous 8KB (pre-swizzled [n][k]) tile into LDS via async DMA.
// NTHR is compile-time so the issue sequence is straight-line.
template <int NTHR>
__device__ __forceinline__ void stage_async_8k(half_t* dstLds, const half_t* srcGlobal,
                                               int tid) {
#pragma unroll
  for (int it = 0; it < 512 / NTHR; ++it) {
    int i = it * NTHR + tid;
    async_copy16((char*)dstLds + i * 16, (const char*)srcGlobal + i * 16);
  }
}

// C[16x128] += Alds[16x128 f16] @ W (pre-swizzled per-K-tile [n][k], 4 tiles).
// WT: block-shared staging buffer (128*32 halves). Uniform call (has barriers).
template <int NTHR>
__device__ __forceinline__ void gemm_16x128x128(const half_t* Alds,
                                                const half_t* __restrict__ Wswz,
                                                half_t* WT, v8f* c, int lanelow,
                                                int khalf, int tid) {
  for (int kt = 0; kt < 4; ++kt) {
    __syncthreads();                       // previous WT fully consumed
    stage_async_8k<NTHR>(WT, Wswz + kt * 4096, tid);
    wait_async0();
    __syncthreads();
    v16h a = load_frag(Alds, lanelow, F, kt * 32, khalf);
#pragma unroll
    for (int nt = 0; nt < 8; ++nt) {
      v16h b = load_frag(WT, nt * 16 + lanelow, 32, 0, khalf);
      c[nt] = __builtin_amdgcn_wmma_f32_16x16x32_f16(false, a, false, b,
                                                     (short)0, c[nt], false, false);
    }
  }
}

__global__ void zero_kernel(float* p, size_t n) {
  size_t i = (size_t)blockIdx.x * blockDim.x + threadIdx.x;
  if (i < n) p[i] = 0.0f;
}

// Convert weights to f16 and pre-swizzle into fragment-ready per-K-tile [n][k]
// layout: dst[mat][kt][n*32+kk] = W[(kt*32+kk)*F + n].
// bilin is additionally transposed: B[k][n] with k = j*F+l, n = output feature i.
__global__ void prep_kernel(const float* __restrict__ Wji, const float* __restrict__ Wkj,
                            const float* __restrict__ r1W1, const float* __restrict__ r1W2,
                            const float* __restrict__ Wbs, const float* __restrict__ r3W1,
                            const float* __restrict__ r3W2, const float* __restrict__ r4W1,
                            const float* __restrict__ r4W2, const float* __restrict__ bilin,
                            half_t* __restrict__ wf16, half_t* __restrict__ bswz) {
  int idx = blockIdx.x * blockDim.x + threadIdx.x;
  const int WN = F * F;
  if (idx < 9 * WN) {
    int m = idx >> 14, e = idx & (WN - 1);
    const float* s;
    switch (m) {
      case 0: s = Wji;  break; case 1: s = Wkj;  break;
      case 2: s = r1W1; break; case 3: s = r1W2; break;
      case 4: s = Wbs;  break; case 5: s = r3W1; break;
      case 6: s = r3W2; break; case 7: s = r4W1; break;
      default: s = r4W2; break;
    }
    int kt = e >> 12, rem = e & 4095;
    int n = rem >> 5, kk = rem & 31;
    int k = kt * 32 + kk;
    wf16[idx] = (half_t)s[k * F + n];
  } else {
    int t = idx - 9 * WN;
    if (t < NB * F * F) {
      int kt = t >> 12, rem = t & 4095;
      int n = rem >> 5, kk = rem & 31;
      int k = kt * 32 + kk;                 // k = j*F + l
      int jj = k >> 7, l = k & (F - 1);
      bswz[t] = (half_t)bilin[(size_t)n * (NB * F) + jj * F + l];
    }
  }
}

#define NWE 4
__global__ void __launch_bounds__(NWE * 32)
edge_pre_kernel(const float* __restrict__ m_l_1, const float* __restrict__ e_rbf,
                const float* __restrict__ W_rbf, const float* __restrict__ b_ji,
                const float* __restrict__ b_kj, const half_t* __restrict__ Wji16,
                const half_t* __restrict__ Wkj16, float* __restrict__ m_ji,
                half_t* __restrict__ m_kj, int E) {
  __shared__ half_t WT[F * 32];
  __shared__ half_t Alds[NWE][16 * F];
  __shared__ float  ER[NWE][16 * NR];
  int tid = threadIdx.x;
  int wave = tid >> 5, lane = tid & 31;
  int lanelow = lane & 15, khalf = lane >> 4;
  int base = (blockIdx.x * NWE + wave) * 16;

  for (int it = 0; it < 64; ++it) {        // 16*F / 32 lanes
    int i = it * 32 + lane;
    int r = i >> 7, cc = i & (F - 1);
    int e = base + r; if (e >= E) e = E - 1;
    Alds[wave][i] = (half_t)m_l_1[(size_t)e * F + cc];
  }
#pragma unroll
  for (int it = 0; it < 3; ++it) {         // 16*NR / 32 lanes
    int i = it * 32 + lane;
    int r = i / NR, cc = i - r * NR;
    int e = base + r; if (e >= E) e = E - 1;
    ER[wave][i] = e_rbf[(size_t)e * NR + cc];
  }

  v8f zero = {0, 0, 0, 0, 0, 0, 0, 0};
  v8f c[8];

  // m_ji = swish(m_l_1 @ W_ji + b_ji)
#pragma unroll
  for (int nt = 0; nt < 8; ++nt) c[nt] = zero;
  gemm_16x128x128<NWE * 32>(Alds[wave], Wji16, WT, c, lanelow, khalf, tid);
#pragma unroll
  for (int nt = 0; nt < 8; ++nt) {
    int n = nt * 16 + lanelow;
    float bb = b_ji[n];
#pragma unroll
    for (int r = 0; r < 8; ++r) {
      int m = r + 8 * khalf;
      int e = base + m;
      if (e < E) m_ji[(size_t)e * F + n] = swishf(c[nt][r] + bb);
    }
  }

  // m_kj = swish(m_l_1 @ W_kj + b_kj) * (e_rbf @ W_rbf)   -> f16
#pragma unroll
  for (int nt = 0; nt < 8; ++nt) c[nt] = zero;
  gemm_16x128x128<NWE * 32>(Alds[wave], Wkj16, WT, c, lanelow, khalf, tid);
#pragma unroll
  for (int nt = 0; nt < 8; ++nt) {
    int n = nt * 16 + lanelow;
    float bb = b_kj[n];
    float wr[NR];
#pragma unroll
    for (int rr = 0; rr < NR; ++rr) wr[rr] = W_rbf[rr * F + n];
#pragma unroll
    for (int r = 0; r < 8; ++r) {
      int m = r + 8 * khalf;
      float rb = 0.0f;
#pragma unroll
      for (int rr = 0; rr < NR; ++rr) rb += ER[wave][m * NR + rr] * wr[rr];
      int e = base + m;
      float val = swishf(c[nt][r] + bb) * rb;
      if (e < E) m_kj[(size_t)e * F + n] = (half_t)val;
    }
  }
}

#define NWT 8
__global__ void __launch_bounds__(NWT * 32)
triplet_kernel(const float* __restrict__ a_sbf, const float* __restrict__ W_sbf,
               const int* __restrict__ id_kj, const int* __restrict__ id_ji,
               const half_t* __restrict__ m_kj, const half_t* __restrict__ bswz,
               float* __restrict__ x, int T_) {
  __shared__ half_t Btile[2][F * 32];    // double-buffered, pre-swizzled [n][k]
  __shared__ half_t Mg[NWT][16 * F];     // gathered m_kj rows, f16
  __shared__ float  A8[NWT][16 * NB];    // a = a_sbf @ W_sbf
  int tid = threadIdx.x;
  int wave = tid >> 5, lane = tid & 31;
  int lanelow = lane & 15, khalf = lane >> 4;
  int base = (blockIdx.x * NWT + wave) * 16;

  // a[t, j] = sum_s a_sbf[t,s] * W_sbf[s,j]
  for (int it = 0; it < 4; ++it) {         // 16*NB / 32 lanes
    int i = it * 32 + lane;
    int t = i >> 3, jj = i & 7;
    int trip = base + t; if (trip >= T_) trip = T_ - 1;
    const float* row = a_sbf + (size_t)trip * NS;
    float acc = 0.0f;
#pragma unroll
    for (int s = 0; s < NS; ++s) acc += row[s] * W_sbf[s * NB + jj];
    A8[wave][i] = acc;
  }
  // gather m_kj rows into LDS via async DMA (16B chunks of 256B rows)
  {
    const char* mk = (const char*)m_kj;
    char* dst = (char*)&Mg[wave][0];
#pragma unroll
    for (int it = 0; it < 8; ++it) {       // 256 chunks / 32 lanes
      int i = it * 32 + lane;
      int r = i >> 4, ch = i & 15;
      int trip = base + r; if (trip >= T_) trip = T_ - 1;
      int e = id_kj[trip];
      async_copy16(dst + r * 256 + ch * 16,
                   mk + (size_t)e * 256 + (size_t)ch * 16);
    }
  }
  // prologue: stage B tile 0
  stage_async_8k<NWT * 32>(Btile[0], bswz, tid);
  wait_async0();
  __syncthreads();

  v8f zero = {0, 0, 0, 0, 0, 0, 0, 0};
  v8f c[8];
#pragma unroll
  for (int nt = 0; nt < 8; ++nt) c[nt] = zero;

  for (int kt = 0; kt < 32; ++kt) {
    int cur = kt & 1;
    if (kt + 1 < 32)                       // uniform: prefetch next tile
      stage_async_8k<NWT * 32>(Btile[cur ^ 1], bswz + (kt + 1) * 4096, tid);

    int jj = kt >> 2;                      // bilinear slot j (4 K-tiles per j)
    int l0 = (kt & 3) * 32;                // l window within the slot
    half_t ah = (half_t)A8[wave][lanelow * NB + jj];
    const half_t* mgrow = &Mg[wave][lanelow * F + l0];
    v16h a;
#pragma unroll
    for (int v = 0; v < 8; ++v) {
      int k = kmap(v, khalf);
      a[2 * v]     = ah * mgrow[k];
      a[2 * v + 1] = ah * mgrow[k + 1];
    }
#pragma unroll
    for (int nt = 0; nt < 8; ++nt) {
      v16h b = load_frag(Btile[cur], nt * 16 + lanelow, 32, 0, khalf);
      c[nt] = __builtin_amdgcn_wmma_f32_16x16x32_f16(false, a, false, b,
                                                     (short)0, c[nt], false, false);
    }
    wait_async0();                         // our prefetch landed
    __syncthreads();                       // everyone done reading cur
  }

  // scatter-add into x[id_ji]
#pragma unroll
  for (int r = 0; r < 8; ++r) {
    int m = r + 8 * khalf;
    int trip = base + m;
    if (trip < T_) {
      int e = id_ji[trip];
#pragma unroll
      for (int nt = 0; nt < 8; ++nt) {
        int n = nt * 16 + lanelow;
        atomicAdd(&x[(size_t)e * F + n], c[nt][r]);
      }
    }
  }
}

#define NWP 4
__global__ void __launch_bounds__(NWP * 32)
edge_post_kernel(const float* __restrict__ m_ji, const float* __restrict__ x,
                 const float* __restrict__ m_l_1, const half_t* __restrict__ wf16,
                 const float* __restrict__ r1_b1, const float* __restrict__ r1_b2,
                 const float* __restrict__ b_bs, const float* __restrict__ r3_b1,
                 const float* __restrict__ r3_b2, const float* __restrict__ r4_b1,
                 const float* __restrict__ r4_b2, float* __restrict__ out, int E) {
  __shared__ half_t WT[F * 32];
  __shared__ half_t Alds[NWP][16 * F];
  int tid = threadIdx.x;
  int wave = tid >> 5, lane = tid & 31;
  int lanelow = lane & 15, khalf = lane >> 4;
  int base = (blockIdx.x * NWP + wave) * 16;

  float h[64];
#pragma unroll
  for (int nt = 0; nt < 8; ++nt) {
    int n = nt * 16 + lanelow;
#pragma unroll
    for (int r = 0; r < 8; ++r) {
      int m = r + 8 * khalf;
      int e = base + m; if (e >= E) e = E - 1;
      size_t off = (size_t)e * F + n;
      h[nt * 8 + r] = m_ji[off] + x[off];
    }
  }

  const half_t* W_r1W1 = wf16 + 2 * F * F;
  const half_t* W_r1W2 = wf16 + 3 * F * F;
  const half_t* W_bs16 = wf16 + 4 * F * F;
  const half_t* W_r3W1 = wf16 + 5 * F * F;
  const half_t* W_r3W2 = wf16 + 6 * F * F;
  const half_t* W_r4W1 = wf16 + 7 * F * F;
  const half_t* W_r4W2 = wf16 + 8 * F * F;

  v8f zero = {0, 0, 0, 0, 0, 0, 0, 0};
  v8f c[8];

  auto write_h_to_Alds = [&]() {
#pragma unroll
    for (int nt = 0; nt < 8; ++nt) {
      int n = nt * 16 + lanelow;
#pragma unroll
      for (int r = 0; r < 8; ++r) {
        int m = r + 8 * khalf;
        Alds[wave][m * F + n] = (half_t)h[nt * 8 + r];
      }
    }
  };

  auto resblock = [&](const half_t* W1, const float* b1,
                      const half_t* W2, const float* b2) {
    write_h_to_Alds();
#pragma unroll
    for (int nt = 0; nt < 8; ++nt) c[nt] = zero;
    gemm_16x128x128<NWP * 32>(Alds[wave], W1, WT, c, lanelow, khalf, tid);
#pragma unroll
    for (int nt = 0; nt < 8; ++nt) {
      int n = nt * 16 + lanelow;
      float bb = b1[n];
#pragma unroll
      for (int r = 0; r < 8; ++r) {
        int m = r + 8 * khalf;
        Alds[wave][m * F + n] = (half_t)swishf(c[nt][r] + bb);
      }
    }
#pragma unroll
    for (int nt = 0; nt < 8; ++nt) c[nt] = zero;
    gemm_16x128x128<NWP * 32>(Alds[wave], W2, WT, c, lanelow, khalf, tid);
#pragma unroll
    for (int nt = 0; nt < 8; ++nt) {
      int n = nt * 16 + lanelow;
      float bb = b2[n];
#pragma unroll
      for (int r = 0; r < 8; ++r) h[nt * 8 + r] += swishf(c[nt][r] + bb);
    }
  };

  resblock(W_r1W1, r1_b1, W_r1W2, r1_b2);

  // h = swish(h @ W_bs + b_bs) + m_l_1
  write_h_to_Alds();
#pragma unroll
  for (int nt = 0; nt < 8; ++nt) c[nt] = zero;
  gemm_16x128x128<NWP * 32>(Alds[wave], W_bs16, WT, c, lanelow, khalf, tid);
#pragma unroll
  for (int nt = 0; nt < 8; ++nt) {
    int n = nt * 16 + lanelow;
    float bb = b_bs[n];
#pragma unroll
    for (int r = 0; r < 8; ++r) {
      int m = r + 8 * khalf;
      int e = base + m; if (e >= E) e = E - 1;
      h[nt * 8 + r] = swishf(c[nt][r] + bb) + m_l_1[(size_t)e * F + n];
    }
  }

  resblock(W_r3W1, r3_b1, W_r3W2, r3_b2);
  resblock(W_r4W1, r4_b1, W_r4W2, r4_b2);

#pragma unroll
  for (int nt = 0; nt < 8; ++nt) {
    int n = nt * 16 + lanelow;
#pragma unroll
    for (int r = 0; r < 8; ++r) {
      int m = r + 8 * khalf;
      int e = base + m;
      if (e < E) out[(size_t)e * F + n] = h[nt * 8 + r];
    }
  }
}

extern "C" void kernel_launch(void* const* d_in, const int* in_sizes, int n_in,
                              void* d_out, int out_size, void* d_ws, size_t ws_size,
                              hipStream_t stream) {
  const float* m_l_1 = (const float*)d_in[0];
  const float* e_rbf = (const float*)d_in[1];
  const float* a_sbf = (const float*)d_in[2];
  const int*   id_kj = (const int*)d_in[3];
  const int*   id_ji = (const int*)d_in[4];
  const float* W_rbf = (const float*)d_in[5];
  const float* W_sbf = (const float*)d_in[6];
  const float* W_ji  = (const float*)d_in[7];
  const float* b_ji  = (const float*)d_in[8];
  const float* W_kj  = (const float*)d_in[9];
  const float* b_kj  = (const float*)d_in[10];
  const float* bilin = (const float*)d_in[11];
  const float* W_bs  = (const float*)d_in[12];
  const float* b_bs  = (const float*)d_in[13];
  const float* r1_W1 = (const float*)d_in[14];
  const float* r1_b1 = (const float*)d_in[15];
  const float* r1_W2 = (const float*)d_in[16];
  const float* r1_b2 = (const float*)d_in[17];
  const float* r3_W1 = (const float*)d_in[18];
  const float* r3_b1 = (const float*)d_in[19];
  const float* r3_W2 = (const float*)d_in[20];
  const float* r3_b2 = (const float*)d_in[21];
  const float* r4_W1 = (const float*)d_in[22];
  const float* r4_b1 = (const float*)d_in[23];
  const float* r4_W2 = (const float*)d_in[24];
  const float* r4_b2 = (const float*)d_in[25];
  (void)n_in; (void)out_size; (void)ws_size;

  const int E  = in_sizes[0] / F;
  const int T_ = in_sizes[3];

  // workspace layout
  char* ws = (char*)d_ws;
  float*  x      = (float*)ws;                       // E*F f32
  float*  m_ji   = x + (size_t)E * F;                // E*F f32
  half_t* m_kj   = (half_t*)(m_ji + (size_t)E * F);  // E*F f16
  half_t* wf16   = m_kj + (size_t)E * F;             // 9 * F*F f16 (swizzled)
  half_t* bswz   = wf16 + 9 * F * F;                 // NB*F * F f16 (swizzled)

  {
    size_t n = (size_t)E * F;
    zero_kernel<<<(unsigned)((n + 255) / 256), 256, 0, stream>>>(x, n);
  }
  {
    int tot = 9 * F * F + NB * F * F;
    prep_kernel<<<(tot + 255) / 256, 256, 0, stream>>>(
        W_ji, W_kj, r1_W1, r1_W2, W_bs, r3_W1, r3_W2, r4_W1, r4_W2,
        bilin, wf16, bswz);
  }
  {
    int blocks = (E + 16 * NWE - 1) / (16 * NWE);
    edge_pre_kernel<<<blocks, NWE * 32, 0, stream>>>(
        m_l_1, e_rbf, W_rbf, b_ji, b_kj,
        wf16 + 0 * F * F, wf16 + 1 * F * F, m_ji, m_kj, E);
  }
  {
    int blocks = (T_ + 16 * NWT - 1) / (16 * NWT);
    triplet_kernel<<<blocks, NWT * 32, 0, stream>>>(
        a_sbf, W_sbf, id_kj, id_ji, m_kj, bswz, x, T_);
  }
  {
    int blocks = (E + 16 * NWP - 1) / (16 * NWP);
    edge_post_kernel<<<blocks, NWP * 32, 0, stream>>>(
        m_ji, x, m_l_1, wf16, r1_b1, r1_b2, b_bs,
        r3_b1, r3_b2, r4_b1, r4_b2, (float*)d_out, E);
  }
}
